// DirectedMultiheadAttention_90598040142501
// MI455X (gfx1250) — matrix operand order repaired and checked by reading, not compiled, not run on hardware
//
#include <hip/hip_runtime.h>
#include <hip/hip_bf16.h>
#include <math.h>

#define EMBED_DIM 1024
#define NUM_HEADS 16
#define HEAD_DIM  64
#define MAX_POS   512
#define TOPK      32
#define BATCH     8
#define SEQ       512
#define QB        32   // query rows per attention block (scores tile = 32x512 fp32 = 64KB LDS)

typedef __attribute__((ext_vector_type(16))) _Float16 v16h;
typedef __attribute__((ext_vector_type(8)))  _Float16 v8h;
typedef __attribute__((ext_vector_type(8)))  float    v8f;

// ---------------------------------------------------------------------------
// WMMA fragment helpers (wave32); A and B use the same contiguous k order, not the ISA table order.
// A (16x32 f16): lanes 0-15 hold K[0..7] (V0-3) and K[8..15] (V4-7) of row M=lane;
//                lanes 16-31 hold K[16..23] and K[24..31] of row M=lane-16.
// B (32x16 f16): lane n holds column N=lane&15, K = (lane>=16 ? 16 : 0)..+15 contiguous.
// C/D (16x16 f32): VGPR i holds (M = i + 8*(lane>=16), N = lane&15).
// ---------------------------------------------------------------------------
__device__ __forceinline__ v16h load_a_frag(const _Float16* base, int ld) {
  const int lane = threadIdx.x & 31;
  const int row  = lane & 15;
  const int hi   = lane >> 4;
  const _Float16* p0 = base + (size_t)row * ld + hi * 16;  // K chunk 0
  const _Float16* p1 = p0 + 8;                             // K chunk 1
  v8h lo = *(const v8h*)p0;
  v8h hh = *(const v8h*)p1;
  v16h r;
#pragma unroll
  for (int i = 0; i < 8; ++i) { r[i] = lo[i]; r[8 + i] = hh[i]; }
  return r;
}

// base points at element (k=0, n=0) of a 32x16 B tile stored so that a column
// (fixed n, varying k) is contiguous with leading dimension `ld`.
__device__ __forceinline__ v16h load_b_frag(const _Float16* base, int ld) {
  const int lane = threadIdx.x & 31;
  const int col  = lane & 15;
  const int hi   = lane >> 4;
  return *(const v16h*)(base + (size_t)col * ld + hi * 16);
}

__device__ __forceinline__ v8f wmma32(v16h a, v16h b, v8f c) {
  return __builtin_amdgcn_wmma_f32_16x16x32_f16(false, a, false, b, (short)0, c,
                                                false, false);
}

// load all 6 fragments (4 A-tiles x 2 B-tiles) for one 32-wide k-slice
__device__ __forceinline__ void load_frags(const _Float16* Abase,
                                           const _Float16* Wbase, int kk,
                                           v16h a[4], v16h b[2]) {
#pragma unroll
  for (int i = 0; i < 4; ++i)
    a[i] = load_a_frag(Abase + (size_t)(16 * i) * EMBED_DIM + kk, EMBED_DIM);
#pragma unroll
  for (int j = 0; j < 2; ++j)
    b[j] = load_b_frag(Wbase + (size_t)(16 * j) * EMBED_DIM + kk, EMBED_DIM);
}

__device__ __forceinline__ void mma_all(v8f acc[4][2], const v16h a[4],
                                        const v16h b[2]) {
#pragma unroll
  for (int i = 0; i < 4; ++i)
#pragma unroll
    for (int j = 0; j < 2; ++j)
      acc[i][j] = wmma32(a[i], b[j], acc[i][j]);
}

// ---------------------------------------------------------------------------
// fp32 -> f16 conversion (grid-stride)
// ---------------------------------------------------------------------------
__global__ void f32_to_f16_kernel(const float* __restrict__ src,
                                  _Float16* __restrict__ dst, int n) {
  int i = blockIdx.x * blockDim.x + threadIdx.x;
  int stride = gridDim.x * blockDim.x;
  for (; i < n; i += stride) dst[i] = (_Float16)src[i];
}

// ---------------------------------------------------------------------------
// C[m,n] = sum_k A[m,k] * W[n,k] + bias[n]   (A: MxK row-major, W: NxK row-major
// so B-matrix columns are W rows -> K-contiguous fragment loads)
// mode 0: store f16, q/k layout   out[((b*H+h)*T + t)*D + d]
// mode 1: store f16, v^T layout   out[((b*H+h)*D + d)*T + t]
// mode 2: store f32 row-major     out[m*EMBED_DIM + n]
// Block: 256 threads (8 waves, 2x4), block tile 128x128, wave tile 64x32.
// k-loop unrolled x2 with ping-pong fragment buffers: no copy/rotate moves,
// each load clause overlapped by a full 8-WMMA window.
// ---------------------------------------------------------------------------
__global__ __launch_bounds__(256) void gemm_wmma_kernel(
    const _Float16* __restrict__ A, const _Float16* __restrict__ W,
    const float* __restrict__ bias, _Float16* __restrict__ outh,
    float* __restrict__ outf, int mode) {
  const int w    = threadIdx.x >> 5;
  const int lane = threadIdx.x & 31;
  const int m0 = blockIdx.x * 128 + (w >> 2) * 64;   // 4 M-tiles per wave
  const int n0 = blockIdx.y * 128 + (w & 3)  * 32;   // 2 N-tiles per wave
  const _Float16* Abase = A + (size_t)m0 * EMBED_DIM;
  const _Float16* Wbase = W + (size_t)n0 * EMBED_DIM;

  v8f acc[4][2] = {};
  v16h a0[4], b0[2], a1[4], b1[2];
  load_frags(Abase, Wbase, 0, a0, b0);
  for (int kk = 0; kk + 64 < EMBED_DIM; kk += 64) {
    load_frags(Abase, Wbase, kk + 32, a1, b1);
    __builtin_prefetch(Abase + kk + 128, 0, 3);   // global_prefetch_b8
    __builtin_prefetch(Wbase + kk + 128, 0, 3);
    mma_all(acc, a0, b0);
    load_frags(Abase, Wbase, kk + 64, a0, b0);
    mma_all(acc, a1, b1);
  }
  // tail: a0 holds k = EMBED_DIM-64; fetch last slice and drain
  load_frags(Abase, Wbase, EMBED_DIM - 32, a1, b1);
  mma_all(acc, a0, b0);
  mma_all(acc, a1, b1);

  const int nlane = lane & 15;
  const int mbase = (lane >> 4) * 8;
#pragma unroll
  for (int tm = 0; tm < 4; ++tm)
#pragma unroll
    for (int tn = 0; tn < 2; ++tn) {
#pragma unroll
      for (int i = 0; i < 8; ++i) {
        int m = m0 + tm * 16 + mbase + i;
        int n = n0 + tn * 16 + nlane;
        float val = acc[tm][tn][i] + bias[n];
        if (mode == 2) {
          outf[(size_t)m * EMBED_DIM + n] = val;
        } else {
          int bi = m >> 9, t = m & 511;
          int h  = n >> 6, d = n & 63;
          if (mode == 0)
            outh[((size_t)(bi * NUM_HEADS + h) * SEQ + t) * HEAD_DIM + d] =
                (_Float16)val;
          else
            outh[((size_t)(bi * NUM_HEADS + h) * HEAD_DIM + d) * SEQ + t] =
                (_Float16)val;
        }
      }
    }
}

// ---------------------------------------------------------------------------
// Fused attention core per (b, h, 32-query block):
//   scores = q k^T / 8 + rel_bias  (WMMA, into LDS fp32)
//   per row: exact 32nd-largest via bitwise radix-select, masked softmax, gate
//   (weights converted to f16 in place), then out = weights @ V (WMMA).
// ---------------------------------------------------------------------------
__global__ __launch_bounds__(256) void attn_kernel(
    const _Float16* __restrict__ q, const _Float16* __restrict__ k,
    const _Float16* __restrict__ vt, const float* __restrict__ rel_bias,
    const float* __restrict__ gates, _Float16* __restrict__ attn_out) {
  __shared__ __align__(128) float sS[QB * SEQ];  // 64 KB of the 320 KB WGP LDS
  const int w    = threadIdx.x >> 5;
  const int lane = threadIdx.x & 31;
  const int qt = blockIdx.x, h = blockIdx.y, bi = blockIdx.z;
  const int bh = bi * NUM_HEADS + h;
  const int qbase = qt * QB;
  const _Float16* qp = q  + (size_t)bh * SEQ * HEAD_DIM;
  const _Float16* kp = k  + (size_t)bh * SEQ * HEAD_DIM;
  const _Float16* vp = vt + (size_t)bh * HEAD_DIM * SEQ;
  const int nlane = lane & 15;
  const int mbase = (lane >> 4) * 8;

  // ---- Phase 1: score tiles (2 M-tiles x 32 N-tiles; 8 tiles per wave) ----
  for (int t8 = 0; t8 < 8; ++t8) {
    int tile = w * 8 + t8;
    int mt = tile >> 5, nt = tile & 31;
    v8f acc = {};
#pragma unroll
    for (int kk = 0; kk < HEAD_DIM; kk += 32) {
      v16h a = load_a_frag(qp + (size_t)(qbase + mt * 16) * HEAD_DIM + kk, HEAD_DIM);
      v16h b = load_b_frag(kp + (size_t)(nt * 16) * HEAD_DIM + kk, HEAD_DIM);
      acc = wmma32(a, b, acc);
    }
#pragma unroll
    for (int i = 0; i < 8; ++i) {
      int r = mt * 16 + mbase + i;
      int key = nt * 16 + nlane;
      int qg = qbase + r;
      float s = acc[i] * 0.125f +
                rel_bias[(size_t)(key - qg + MAX_POS - 1) * NUM_HEADS + h];
      sS[r * SEQ + key] = s;
    }
  }
  __syncthreads();

  // ---- Phase 2: per-row top-32 threshold + softmax + gate (4 rows/wave) ----
  const float gate = gates[h];
  for (int rr = 0; rr < QB / 8; ++rr) {
    int r = w + rr * 8;
    float f[16];
    unsigned keyb[16];
#pragma unroll
    for (int e = 0; e < 16; ++e) {
      float x = sS[r * SEQ + lane + 32 * e];
      f[e] = x;
      unsigned u = __float_as_uint(x);
      keyb[e] = (u & 0x80000000u) ? ~u : (u | 0x80000000u);  // monotone key
    }
    // exact 32nd-largest via MSB-first radix select on keys
    unsigned prefix = 0;
    int rem = TOPK;
    for (int b = 31; b >= 0; --b) {
      unsigned want = prefix | (1u << b);
      int cnt = 0;
#pragma unroll
      for (int e = 0; e < 16; ++e) cnt += ((keyb[e] >> b) == (want >> b));
#pragma unroll
      for (int off = 16; off > 0; off >>= 1) cnt += __shfl_xor(cnt, off, 32);
      if (cnt >= rem) prefix = want; else rem -= cnt;
    }
    // masked softmax (row max is always among kept elements)
    float mx = -__builtin_inff();
#pragma unroll
    for (int e = 0; e < 16; ++e) mx = fmaxf(mx, f[e]);
#pragma unroll
    for (int off = 16; off > 0; off >>= 1) mx = fmaxf(mx, __shfl_xor(mx, off, 32));
    float ev[16], sum = 0.f;
#pragma unroll
    for (int e = 0; e < 16; ++e) {
      float v = (keyb[e] >= prefix) ? __expf(f[e] - mx) : 0.f;
      ev[e] = v; sum += v;
    }
#pragma unroll
    for (int off = 16; off > 0; off >>= 1) sum += __shfl_xor(sum, off, 32);
    float scale = gate / sum;
    _Float16* hrow = (_Float16*)(sS + r * SEQ);  // f16 weights in place
#pragma unroll
    for (int e = 0; e < 16; ++e) hrow[lane + 32 * e] = (_Float16)(ev[e] * scale);
  }
  __syncthreads();

  // ---- Phase 3: out tile (32x64) = weights(32x512) @ V(512x64), 1/wave ----
  {
    int mt = w >> 2, nt = w & 3;
    v8f acc = {};
    const _Float16* wbase =
        (const _Float16*)sS + (size_t)(mt * 16) * (SEQ * 2);  // LDS weights
    const _Float16* vbase = vp + (size_t)(nt * 16) * SEQ;
    v16h pa0, pb0, pa1, pb1;
    pa0 = load_a_frag(wbase, SEQ * 2);
    pb0 = load_b_frag(vbase, SEQ);
    for (int kk = 0; kk + 64 < SEQ; kk += 64) {
      pa1 = load_a_frag(wbase + kk + 32, SEQ * 2);
      pb1 = load_b_frag(vbase + kk + 32, SEQ);
      acc = wmma32(pa0, pb0, acc);
      pa0 = load_a_frag(wbase + kk + 64, SEQ * 2);
      pb0 = load_b_frag(vbase + kk + 64, SEQ);
      acc = wmma32(pa1, pb1, acc);
    }
    pa1 = load_a_frag(wbase + SEQ - 32, SEQ * 2);
    pb1 = load_b_frag(vbase + SEQ - 32, SEQ);
    acc = wmma32(pa0, pb0, acc);
    acc = wmma32(pa1, pb1, acc);
#pragma unroll
    for (int i = 0; i < 8; ++i) {
      int t = qbase + mt * 16 + mbase + i;
      int c = h * HEAD_DIM + nt * 16 + nlane;
      attn_out[(size_t)(bi * SEQ + t) * EMBED_DIM + c] = (_Float16)acc[i];
    }
  }
}

// ---------------------------------------------------------------------------
extern "C" void kernel_launch(void* const* d_in, const int* in_sizes, int n_in,
                              void* d_out, int out_size, void* d_ws,
                              size_t ws_size, hipStream_t stream) {
  (void)in_sizes; (void)n_in; (void)out_size; (void)ws_size;
  const float* x     = (const float*)d_in[0];
  const float* Wq    = (const float*)d_in[1];
  const float* bq    = (const float*)d_in[2];
  const float* Wk    = (const float*)d_in[3];
  const float* bk    = (const float*)d_in[4];
  const float* Wv    = (const float*)d_in[5];
  const float* bv    = (const float*)d_in[6];
  const float* Wo    = (const float*)d_in[7];
  const float* bo    = (const float*)d_in[8];
  const float* gates = (const float*)d_in[9];
  const float* relb  = (const float*)d_in[10];

  // workspace layout (f16): x | Wq | Wk | Wv | Wo | q | k | v^T | attn  (~48MB)
  const size_t NX = (size_t)BATCH * SEQ * EMBED_DIM;   // 4,194,304
  const size_t NW = (size_t)EMBED_DIM * EMBED_DIM;     // 1,048,576
  _Float16* xh  = (_Float16*)d_ws;
  _Float16* wqh = xh  + NX;
  _Float16* wkh = wqh + NW;
  _Float16* wvh = wkh + NW;
  _Float16* woh = wvh + NW;
  _Float16* qh  = woh + NW;
  _Float16* kh  = qh  + NX;
  _Float16* vth = kh  + NX;
  _Float16* ah  = vth + NX;

  f32_to_f16_kernel<<<1024, 256, 0, stream>>>(x,  xh,  (int)NX);
  f32_to_f16_kernel<<<512,  256, 0, stream>>>(Wq, wqh, (int)NW);
  f32_to_f16_kernel<<<512,  256, 0, stream>>>(Wk, wkh, (int)NW);
  f32_to_f16_kernel<<<512,  256, 0, stream>>>(Wv, wvh, (int)NW);
  f32_to_f16_kernel<<<512,  256, 0, stream>>>(Wo, woh, (int)NW);

  dim3 ggrid(BATCH * SEQ / 128, EMBED_DIM / 128);   // 32 x 8 = 256 blocks
  gemm_wmma_kernel<<<ggrid, 256, 0, stream>>>(xh, wqh, bq, qh,  nullptr, 0);
  gemm_wmma_kernel<<<ggrid, 256, 0, stream>>>(xh, wkh, bk, kh,  nullptr, 0);
  gemm_wmma_kernel<<<ggrid, 256, 0, stream>>>(xh, wvh, bv, vth, nullptr, 1);

  dim3 agrid(SEQ / QB, NUM_HEADS, BATCH);
  attn_kernel<<<agrid, 256, 0, stream>>>(qh, kh, vth, relb, gates, ah);

  gemm_wmma_kernel<<<ggrid, 256, 0, stream>>>(ah, woh, bo, nullptr,
                                              (float*)d_out, 2);
}
